// Performer_67568425500830
// MI455X (gfx1250) — matrix operand order repaired and checked by reading, not compiled, not run on hardware
//
#include <hip/hip_runtime.h>
#include <hip/hip_bf16.h>
#include <math.h>

// ---------------------------------------------------------------------------
// Performer (FAVOR+) attention, fused, for MI455X (gfx1250, wave32, WMMA).
// B=8 H=12 N=4096 D=64 M=256.  All GEMMs use v_wmma_f32_16x16x32_f16.
// proj/context pre-swizzled to f16 WMMA B-fragment layout; V staged to LDS
// with global_load_async_to_lds_b128 (ASYNCcnt) overlapping the WMMA work.
// ---------------------------------------------------------------------------

#define NN   4096
#define DD   64
#define MM   256
#define BHC  96            // B*H
#define NSPLIT 4           // n-range splits for the context kernel

#define NORMALIZER 0.35355339059327373f   // 64^-0.25
#define DIAGC      0.0625f                // 0.5 * 64^-0.5
#define RATIO      0.0625f                // 256^-0.5
#define EPSV       1e-4f

typedef __attribute__((ext_vector_type(16))) _Float16 v16h;
typedef __attribute__((ext_vector_type(8)))  float    v8f;

// ---- WMMA wrapper: D = A(16x32 f16) * B(32x16 f16) + C(16x16 f32) ---------
__device__ __forceinline__ v8f wmma32(v16h a, v16h b, v8f c) {
  return __builtin_amdgcn_wmma_f32_16x16x32_f16(
      /*neg_a=*/false, a, /*neg_b=*/false, b,
      /*c_mod=*/(short)0, c, /*reuse_a=*/false, /*reuse_b=*/false);
}

// ---- async global->LDS DMA (ASYNCcnt) -------------------------------------
__device__ __forceinline__ void async_ld_b128(unsigned lds_byte_off,
                                              const void* gaddr) {
  asm volatile("global_load_async_to_lds_b128 %0, %1, off"
               :: "v"(lds_byte_off),
                  "v"((unsigned long long)(uintptr_t)gaddr)
               : "memory");
}
__device__ __forceinline__ void wait_async0() {
  asm volatile("s_wait_asynccnt 0" ::: "memory");
}

// ---- Fragment loaders (contiguous k order K = 16*(lane>>4) + i, wave32) --------------
// A fragment (16 rows x 32 K) from f32 row-major matrix, scaled, cvt->f16.
// lane: row = base + (lane&15); element i: K = kc*32 + i + (lane>>4)*16
__device__ __forceinline__ v16h load_A_f32(const float* __restrict__ base,
                                           int ld, int row0, int kc,
                                           float scale, unsigned lane) {
  const int r = lane & 15, hi = lane >> 4;
  const float* p = base + (size_t)(row0 + r) * ld + kc * 32 + hi * 16;
  v16h a;
#pragma unroll
  for (int i = 0; i < 8; ++i) a[i]     = (_Float16)(scale * p[i]);
#pragma unroll
  for (int i = 0; i < 8; ++i) a[8 + i] = (_Float16)(scale * p[8 + i]);
  return a;
}

// A fragment from LDS f16 (row-major [row][K], stride ld halves)
__device__ __forceinline__ v16h load_A_lds(const _Float16* __restrict__ lds,
                                           int ld, int row0, int k0,
                                           unsigned lane) {
  const int r = lane & 15, hi = lane >> 4;
  const _Float16* p = lds + (size_t)(row0 + r) * ld + k0 + hi * 16;
  v16h a;
#pragma unroll
  for (int i = 0; i < 8; ++i) a[i]     = p[i];
#pragma unroll
  for (int i = 0; i < 8; ++i) a[8 + i] = p[8 + i];
  return a;
}

// B fragment from pre-swizzled f16 fragment buffer: frag f, lane -> one v16h
__device__ __forceinline__ v16h load_B_frag(const _Float16* __restrict__ fb,
                                            int frag, unsigned lane) {
  return *(const v16h*)(fb + ((size_t)frag * 32 + lane) * 16);
}

// B fragment (32 K x 16 N) from f32 matrix in LDS stored [K][N] (V stage)
__device__ __forceinline__ v16h load_B_lds_f32(const float* __restrict__ lds,
                                               int ld, int k0, int c0,
                                               unsigned lane) {
  const int c = lane & 15, hi = lane >> 4;
  v16h b;
#pragma unroll
  for (int i = 0; i < 16; ++i)
    b[i] = (_Float16)lds[(size_t)(k0 + i + hi * 16) * ld + c0 + c];
  return b;
}

// ---- order-preserving float<->uint map for deterministic atomic max -------
__device__ __forceinline__ unsigned fmap(float f) {
  unsigned b = __float_as_uint(f);
  return (b & 0x80000000u) ? ~b : (b | 0x80000000u);
}
__device__ __forceinline__ float funmap(unsigned u) {
  return __uint_as_float((u & 0x80000000u) ? (u & 0x7FFFFFFFu) : ~u);
}

// ===========================================================================
// Kernel 0a: init global-max cell
// ===========================================================================
__global__ void perf_init(unsigned* kmax_u) {
  if (threadIdx.x == 0 && blockIdx.x == 0) kmax_u[0] = 0u;
}

// ===========================================================================
// Kernel 0b: pre-swizzle proj into f16 B-fragment layout.
// frag index = mt*2 + kc  (mt 0..15, kc 0..1);  per frag: [lane][16 halves].
// Frag element i of lane: K = kc*32 + i + (lane>>4)*16, col = mt*16+(lane&15)
// ===========================================================================
__global__ void perf_projswz(const float* __restrict__ proj,
                             _Float16* __restrict__ projF) {
  for (int idx = threadIdx.x; idx < 16 * 2 * 32; idx += 256) {
    const int lane = idx & 31, kc = (idx >> 5) & 1, mt = idx >> 6;
    const int c = lane & 15, hi = lane >> 4;
    const float* p = proj + (size_t)(mt * 16 + c) * DD + kc * 32 + hi * 16;
    _Float16* o = projF + (size_t)idx * 16;
#pragma unroll
    for (int i = 0; i < 16; ++i) o[i] = (_Float16)p[i];
  }
}

// ===========================================================================
// Kernel 1: global max over S_k = (norm*K) @ projT.   8 waves/block,
// each wave: 16 rows x all 256 features (16 m-tiles), 32 WMMAs.
// ===========================================================================
__global__ void perf_kmax(const float* __restrict__ k,
                          const _Float16* __restrict__ projF,
                          unsigned* __restrict__ kmax_u) {
  const unsigned tid = threadIdx.x, lane = tid & 31, w = tid >> 5;
  const unsigned bh   = blockIdx.x / (NN / 128);
  const unsigned nblk = blockIdx.x % (NN / 128);
  const float* kb = k + (size_t)bh * NN * DD;
  const int nrow = nblk * 128 + (int)w * 16;

  v16h a0 = load_A_f32(kb, DD, nrow, 0, NORMALIZER, lane);
  v16h a1 = load_A_f32(kb, DD, nrow, 1, NORMALIZER, lane);
  float mx = -3.4e38f;
#pragma unroll 4
  for (int mt = 0; mt < 16; ++mt) {
    v16h b0 = load_B_frag(projF, mt * 2 + 0, lane);
    v16h b1 = load_B_frag(projF, mt * 2 + 1, lane);
    v8f c = {};
    c = wmma32(a0, b0, c);
    c = wmma32(a1, b1, c);
#pragma unroll
    for (int i = 0; i < 8; ++i) mx = fmaxf(mx, c[i]);
  }
#pragma unroll
  for (int off = 16; off; off >>= 1) mx = fmaxf(mx, __shfl_xor(mx, off, 32));
  if (lane == 0) atomicMax(kmax_u, fmap(mx));   // order-independent
}

// ===========================================================================
// Kernel 2: fused kp + context + k_sum.
// Per 32-row chunk: async-DMA the V chunk to LDS (overlapped with phase 1),
//   phase1: S = (norm*K)@projT (WMMA), kp = ratio*(exp(S-diag-kmax)+eps),
//           store kp^T (f16) into LDS [m][n_local]; accumulate k_sum cols.
//   phase2: context[m,d] += kp^T @ V_lds  (WMMA, K = 32 chunk rows).
// ===========================================================================
__global__ void perf_kctx(const float* __restrict__ k,
                          const float* __restrict__ v,
                          const _Float16* __restrict__ projF,
                          const unsigned* __restrict__ kmax_u,
                          float* __restrict__ ksumP,
                          float* __restrict__ ctxP) {
  __shared__ float    diag_s[32];
  __shared__ float    vstage[32][DD];    // async-staged V chunk (f32)
  __shared__ _Float16 kpT[MM][40];       // [m][n_local], padded stride
  __shared__ float    ksum_part[8][64];

  const unsigned tid = threadIdx.x, lane = tid & 31, w = tid >> 5;
  const unsigned bh = blockIdx.x / NSPLIT, split = blockIdx.x % NSPLIT;
  const int ROWS = NN / NSPLIT;
  const float* kb = k + (size_t)bh * NN * DD;
  const float* vb = v + (size_t)bh * NN * DD;
  const float kmax = funmap(*kmax_u);

  const int p1_nt = (w >= 4) ? 1 : 0;        // phase-1 n-tile
  const int p1_mb = (int)(w & 3) * 4;        // phase-1 first m-tile
  float ksum_acc[4] = {0.f, 0.f, 0.f, 0.f};
  v8f acc[2][4] = {};                        // phase-2 accumulators

  // per-thread async copy slot: 8 consecutive floats of the 32x64 chunk
  const int vrow = (int)tid >> 3, vcol = ((int)tid & 7) * 8;
  const unsigned vst_off = (unsigned)(uintptr_t)&vstage[vrow][vcol];

  const int nbase0 = split * ROWS;
  for (int chunk = 0; chunk < ROWS / 32; ++chunk) {
    const int n0 = nbase0 + chunk * 32;

    // ---- kick off async DMA of V chunk (32x64 f32) into LDS ----
    {
      const float* g = vb + (size_t)(n0 + vrow) * DD + vcol;
      async_ld_b128(vst_off, g);
      async_ld_b128(vst_off + 16, g + 4);
    }

    if (tid < 32) {
      const float* p = kb + (size_t)(n0 + tid) * DD;
      float s = 0.f;
      for (int d = 0; d < DD; ++d) s += p[d] * p[d];
      diag_s[tid] = DIAGC * s;
    }
    __syncthreads();

    { // ---- phase 1: projection GEMM + feature map -> LDS kp^T ----
      const int nrow = n0 + p1_nt * 16;
      v16h a0 = load_A_f32(kb, DD, nrow, 0, NORMALIZER, lane);
      v16h a1 = load_A_f32(kb, DD, nrow, 1, NORMALIZER, lane);
#pragma unroll
      for (int j = 0; j < 4; ++j) {
        const int mt = p1_mb + j;
        v16h b0 = load_B_frag(projF, mt * 2 + 0, lane);
        v16h b1 = load_B_frag(projF, mt * 2 + 1, lane);
        v8f c = {};
        c = wmma32(a0, b0, c);
        c = wmma32(a1, b1, c);
        const int m = mt * 16 + (lane & 15);
        float colsum = 0.f;
#pragma unroll
        for (int rr = 0; rr < 8; ++rr) {
          const int nl = p1_nt * 16 + rr + (int)(lane >> 4) * 8;
          float val = RATIO * (__expf(c[rr] - diag_s[nl] - kmax) + EPSV);
          colsum += val;
          kpT[m][nl] = (_Float16)val;
        }
        ksum_acc[j] += colsum;
      }
    }
    wait_async0();       // my wave's DMA into vstage complete
    __syncthreads();     // all waves' DMA + kpT writes visible

    { // ---- phase 2: context += kp^T @ V_lds ----
      v16h bf[4];
#pragma unroll
      for (int dt = 0; dt < 4; ++dt)
        bf[dt] = load_B_lds_f32(&vstage[0][0], DD, 0, dt * 16, lane);
#pragma unroll
      for (int i = 0; i < 2; ++i) {
        v16h af = load_A_lds(&kpT[0][0], 40, (2 * (int)w + i) * 16, 0, lane);
#pragma unroll
        for (int dt = 0; dt < 4; ++dt) acc[i][dt] = wmma32(af, bf[dt], acc[i][dt]);
      }
    }
    __syncthreads();     // protect vstage/kpT before next chunk overwrites
  }

  // ---- k_sum: fold halves, stage per-wave, fixed-order combine ----
#pragma unroll
  for (int j = 0; j < 4; ++j) ksum_acc[j] += __shfl_xor(ksum_acc[j], 16, 32);
  if (lane < 16) {
#pragma unroll
    for (int j = 0; j < 4; ++j) ksum_part[w][j * 16 + lane] = ksum_acc[j];
  }
  __syncthreads();
  {
    const int m = tid;
    const int w1 = m >> 6;
    ksumP[((size_t)bh * NSPLIT + split) * MM + m] =
        ksum_part[w1][m & 63] + ksum_part[w1 + 4][m & 63];
  }
  float* cp = ctxP + ((size_t)bh * NSPLIT + split) * MM * DD;
#pragma unroll
  for (int i = 0; i < 2; ++i) {
    const int mt = 2 * (int)w + i;
#pragma unroll
    for (int dt = 0; dt < 4; ++dt) {
#pragma unroll
      for (int rr = 0; rr < 8; ++rr) {
        const int m = mt * 16 + rr + (int)(lane >> 4) * 8;
        const int d = dt * 16 + (lane & 15);
        cp[(size_t)m * DD + d] = acc[i][dt][rr];
      }
    }
  }
}

// ===========================================================================
// Kernel 3: deterministic fixed-order reduce of NSPLIT partials.
// k_sum -> f32; context -> f16 WMMA B-fragment layout (frag = kc*4 + dt).
// ===========================================================================
__global__ void perf_reduce(const float* __restrict__ ksumP,
                            const float* __restrict__ ctxP,
                            float* __restrict__ ksumF,
                            _Float16* __restrict__ ctxFh) {
  const int bh = blockIdx.x;
  for (int m = threadIdx.x; m < MM; m += 256) {
    float s = 0.f;
    for (int p = 0; p < NSPLIT; ++p) s += ksumP[((size_t)bh * NSPLIT + p) * MM + m];
    ksumF[(size_t)bh * MM + m] = s;
  }
  for (int i = threadIdx.x; i < MM * DD; i += 256) {
    const int m = i / DD, d = i % DD;
    float s = 0.f;
    for (int p = 0; p < NSPLIT; ++p) s += ctxP[((size_t)bh * NSPLIT + p) * MM * DD + i];
    // B-fragment position for (K=m, N=d):
    const int kc = m >> 5, rem = m & 31, hi = rem >> 4, e = rem & 15;
    const int dt = d >> 4, lane = hi * 16 + (d & 15);
    ctxFh[(size_t)bh * MM * DD + (((size_t)(kc * 4 + dt) * 32) + lane) * 16 + e] =
        (_Float16)s;
  }
}

// ===========================================================================
// Kernel 4: fused qp + output.  Grid BHC*(N/32) blocks, 256 thr.
// ===========================================================================
__global__ void perf_qout(const float* __restrict__ q,
                          const _Float16* __restrict__ projF,
                          const float* __restrict__ ksumF,
                          const _Float16* __restrict__ ctxFh,
                          float* __restrict__ out) {
  __shared__ float    Sq[32][260];
  __shared__ _Float16 qp_s[32][264];
  __shared__ float    ksum_s[MM];
  __shared__ float    pmax[32][8], pden[32][8];
  __shared__ float    stab[32], dinv[32], diag_s[32];

  const unsigned tid = threadIdx.x, lane = tid & 31, w = tid >> 5;
  const unsigned bh = blockIdx.x / (NN / 32);
  const unsigned nc = blockIdx.x % (NN / 32);
  const int n0 = (int)nc * 32;
  const float* qb = q + (size_t)bh * NN * DD;
  const _Float16* ch = ctxFh + (size_t)bh * MM * DD;

  ksum_s[tid] = ksumF[(size_t)bh * MM + tid];
  if (tid < 32) {
    const float* p = qb + (size_t)(n0 + tid) * DD;
    float s = 0.f;
    for (int d = 0; d < DD; ++d) s += p[d] * p[d];
    diag_s[tid] = DIAGC * s;
  }

  { // ---- phase 1: S_q tiles (WMMA) -> LDS ----
    const int nt = (int)w >> 2;
    v16h a0 = load_A_f32(qb, DD, n0 + nt * 16, 0, NORMALIZER, lane);
    v16h a1 = load_A_f32(qb, DD, n0 + nt * 16, 1, NORMALIZER, lane);
#pragma unroll
    for (int j = 0; j < 4; ++j) {
      const int mt = ((int)w & 3) * 4 + j;
      v16h b0 = load_B_frag(projF, mt * 2 + 0, lane);
      v16h b1 = load_B_frag(projF, mt * 2 + 1, lane);
      v8f c = {};
      c = wmma32(a0, b0, c);
      c = wmma32(a1, b1, c);
      const int m = mt * 16 + (lane & 15);
#pragma unroll
      for (int rr = 0; rr < 8; ++rr)
        Sq[nt * 16 + rr + (int)(lane >> 4) * 8][m] = c[rr];
    }
  }
  __syncthreads();

  { // ---- phase 2a: per-row max partials ----
    const int row = tid >> 3, seg = tid & 7;
    float mx = -3.4e38f;
    for (int m = seg * 32; m < seg * 32 + 32; ++m) mx = fmaxf(mx, Sq[row][m]);
    pmax[row][seg] = mx;
  }
  __syncthreads();
  if (tid < 32) {
    float mx = pmax[tid][0];
    for (int s = 1; s < 8; ++s) mx = fmaxf(mx, pmax[tid][s]);
    stab[tid] = mx;
  }
  __syncthreads();
  { // ---- phase 2b: exp + denominator partial ----
    const int row = tid >> 3, seg = tid & 7;
    const float dg = diag_s[row], st = stab[row];
    float den = 0.f;
    for (int m = seg * 32; m < seg * 32 + 32; ++m) {
      float val = RATIO * (__expf(Sq[row][m] - dg - st) + EPSV);
      qp_s[row][m] = (_Float16)val;
      den += val * ksum_s[m];
    }
    pden[row][seg] = den;
  }
  __syncthreads();
  if (tid < 32) {
    float den = pden[tid][0];
    for (int s = 1; s < 8; ++s) den += pden[tid][s];
    dinv[tid] = 1.0f / den;
  }
  __syncthreads();

  { // ---- phase 3: out = (qp @ ctx) * dinv ----
    const int nt = (int)w >> 2, dt = (int)w & 3;
    v8f c = {};
#pragma unroll
    for (int kc = 0; kc < 8; ++kc) {
      v16h a = load_A_lds(&qp_s[0][0], 264, nt * 16, kc * 32, lane);
      v16h b = load_B_frag(ch, kc * 4 + dt, lane);
      c = wmma32(a, b, c);
    }
    float* ob = out + (size_t)bh * NN * DD;
#pragma unroll
    for (int rr = 0; rr < 8; ++rr) {
      const int nl = nt * 16 + rr + (int)(lane >> 4) * 8;
      ob[(size_t)(n0 + nl) * DD + dt * 16 + (lane & 15)] = c[rr] * dinv[nl];
    }
  }
}

// ===========================================================================
extern "C" void kernel_launch(void* const* d_in, const int* in_sizes, int n_in,
                              void* d_out, int out_size, void* d_ws, size_t ws_size,
                              hipStream_t stream) {
  (void)in_sizes; (void)n_in; (void)out_size; (void)ws_size;
  const float* q    = (const float*)d_in[0];
  const float* k    = (const float*)d_in[1];
  const float* v    = (const float*)d_in[2];
  const float* proj = (const float*)d_in[3];
  float* out = (float*)d_out;

  // workspace layout (~29 MB), all regions 32B-aligned:
  float*    ws     = (float*)d_ws;
  unsigned* kmax_u = (unsigned*)d_ws;                           // 1 uint
  float* ksumP = ws + 64;                                       // 96*4*256
  float* ctxP  = ksumP + (size_t)BHC * NSPLIT * MM;             // 96*4*256*64
  float* ksumF = ctxP + (size_t)BHC * NSPLIT * MM * DD;         // 96*256
  _Float16* ctxFh = (_Float16*)(ksumF + (size_t)BHC * MM);      // 96*256*64 f16
  _Float16* projF = ctxFh + (size_t)BHC * MM * DD;              // 256*64 f16

  perf_init   <<<1, 64, 0, stream>>>(kmax_u);
  perf_projswz<<<1, 256, 0, stream>>>(proj, projF);
  perf_kmax   <<<BHC * (NN / 128), 256, 0, stream>>>(k, projF, kmax_u);
  perf_kctx   <<<BHC * NSPLIT,     256, 0, stream>>>(k, v, projF, kmax_u, ksumP, ctxP);
  perf_reduce <<<BHC,              256, 0, stream>>>(ksumP, ctxP, ksumF, ctxFh);
  perf_qout   <<<BHC * (NN / 32),  256, 0, stream>>>(q, projF, ksumF, ctxFh, out);
}
